// FullModel_21337397526699
// MI455X (gfx1250) — matrix operand order repaired and checked by reading, not compiled, not run on hardware
//
#include <hip/hip_runtime.h>

typedef __attribute__((ext_vector_type(16))) _Float16 v16h;
typedef __attribute__((ext_vector_type(8)))  _Float16 v8h;
typedef __attribute__((ext_vector_type(8)))  float    v8f;

#define NN 100000      // node_size
#define RR 1000        // rel_size
#define TT 500         // time_size
#define EE 1000000     // triple_size
#define DD 50          // emb_size
#define NNZ_E 500000
#define NNZ_R 50000
#define NNZ_T 25000
#define OC 200         // output cols = (DEPTH+2)*D

static __device__ __forceinline__ void atomAddF(float* p, float v) {
  __hip_atomic_fetch_add(p, v, __ATOMIC_RELAXED, __HIP_MEMORY_SCOPE_AGENT);
}
// monotone float <-> ordered-int mapping for atomicMax on floats
static __device__ __forceinline__ int f2ord(float f) {
  int i = __float_as_int(f);
  return (i >= 0) ? i : (i ^ 0x7FFFFFFF);
}
static __device__ __forceinline__ float ord2f(int k) {
  return __int_as_float((k >= 0) ? k : (k ^ 0x7FFFFFFF));
}
#define ORD_NEG_INF 0x807FFFFF   // f2ord(-inf)

__global__ void k_zero(float* p, int n) {
  int i = blockIdx.x * blockDim.x + threadIdx.x;
  if (i < n) p[i] = 0.0f;
}

__global__ void k_count(const int* rows, int nnz, int* cnt) {
  int i = blockIdx.x * blockDim.x + threadIdx.x;
  if (i < nnz) atomicAdd(&cnt[rows[i]], 1);
}

// ctx_emb scatter: emb[rows[j]] += table[cols[j]]   (divide by count later)
__global__ void k_ctx_scatter(const int* rows, const int* cols, const float* table,
                              float* emb, int nnz) {
  int i = blockIdx.x * blockDim.x + threadIdx.x;
  if (i < nnz * DD) {
    int j = i / DD, d = i - j * DD;
    atomAddF(&emb[(long)rows[j] * DD + d], table[(long)cols[j] * DD + d]);
  }
}

__global__ void k_ctx_div(float* emb, const int* cnt, int nrows) {
  int i = blockIdx.x * blockDim.x + threadIdx.x;
  if (i < nrows * DD) {
    int r = i / DD;
    int c = cnt[r];
    if (c > 0) emb[i] *= (1.0f / (float)c);
  }
}

// buf[edge[j]] += val[j] * emb[ref[j]]   over j in [0,E)
__global__ void k_edge_scatter(const int* edge, const int* ref, const float* val,
                               const float* emb, float* buf) {
  int i = blockIdx.x * blockDim.x + threadIdx.x;
  if (i < EE * DD) {
    int j = i / DD, d = i - j * DD;
    atomAddF(&buf[(long)edge[j] * DD + d], val[j] * emb[(long)ref[j] * DD + d]);
  }
}

// one wave per edge: dot3[e] = dot(buf[e], w[100:150]) / max(||buf[e]||, eps)
// optionally scatter buf[e] into out[adj_src[e], 150:200] (time encoding)
__global__ void k_norm_dot(const float* buf, const float* w150, float* dot3,
                           int do_scatter, const int* adj_src, float* out) {
  int wid  = (blockIdx.x * blockDim.x + threadIdx.x) >> 5;
  int lane = threadIdx.x & 31;
  if (wid >= EE) return;
  const float* row = buf + (long)wid * DD;
  float v0 = (lane < DD) ? row[lane] : 0.f;
  float v1 = (lane + 32 < DD) ? row[lane + 32] : 0.f;
  float w0 = (lane < DD) ? w150[100 + lane] : 0.f;
  float w1 = (lane + 32 < DD) ? w150[100 + lane + 32] : 0.f;
  float n2 = v0 * v0 + v1 * v1;
  float dt = v0 * w0 + v1 * w1;
  for (int s = 16; s > 0; s >>= 1) {
    n2 += __shfl_down(n2, s, 32);
    dt += __shfl_down(dt, s, 32);
  }
  if (lane == 0) dot3[wid] = dt / fmaxf(sqrtf(n2), 1e-12f);
  if (do_scatter) {
    int src = adj_src[wid];
    float* o = out + (long)src * OC + 3 * DD;
    if (lane < DD)      atomAddF(&o[lane], v0);
    if (lane + 32 < DD) atomAddF(&o[lane + 32], v1);
  }
}

// 16-element f16 fragment assembled from two aligned 16B LDS vectors
union H16 {
  v16h v;
  struct { v8h lo, hi; } p;
};

// WMMA logit kernel: 16 edges per wave. A = [selfs|neighs] (16x100 -> padded 16x128 f16),
// B col0 = w_rel[0:100], col1 = w_time[0:100], cols 2..15 = zeros.
// 4x v_wmma_f32_16x16x32_f16 accumulate; all LDS traffic is ds_load_b128, branch-free.
__global__ void __launch_bounds__(32) k_logits_wmma(
    const float* __restrict__ ent_emb, const int* __restrict__ adj_src,
    const int* __restrict__ adj_dst,
    const float* __restrict__ wrel, const float* __restrict__ wtime,
    const float* __restrict__ rel_dot3, const float* __restrict__ time_dot3,
    const float* __restrict__ b_rel, const float* __restrict__ b_time,
    float* __restrict__ rel_logit, float* __restrict__ time_logit) {
  __shared__ _Float16 A_s[16][128];
  __shared__ _Float16 Wr_s[128];
  __shared__ _Float16 Wt_s[128];
  __shared__ _Float16 Z_s[128];
  __shared__ int idx_s[32];
  const int lane  = threadIdx.x;
  const int ebase = blockIdx.x * 16;

  if (lane < 16) idx_s[lane] = adj_src[ebase + lane];
  else           idx_s[lane] = adj_dst[ebase + (lane - 16)];
  for (int i = lane; i < 128; i += 32) {
    Wr_s[i] = (i < 100) ? (_Float16)wrel[i]  : (_Float16)0.f;
    Wt_s[i] = (i < 100) ? (_Float16)wtime[i] : (_Float16)0.f;
    Z_s[i]  = (_Float16)0.f;
  }
  __syncthreads();
  // stage selfs (cols 0..49), neighs (cols 50..99), pad (cols 100..127) -- no inner branches
  for (int i = lane; i < 16 * DD; i += 32) {
    int r = i / DD, c = i - r * DD;
    A_s[r][c] = (_Float16)ent_emb[(long)idx_s[r] * DD + c];
  }
  for (int i = lane; i < 16 * DD; i += 32) {
    int r = i / DD, c = i - r * DD;
    A_s[r][DD + c] = (_Float16)ent_emb[(long)idx_s[16 + r] * DD + c];
  }
  for (int i = lane; i < 16 * 28; i += 32) {
    int r = i / 28, c = i - r * 28;
    A_s[r][100 + c] = (_Float16)0.f;
  }
  __syncthreads();

  const int r  = lane & 15;        // A: row; B/D: column
  const int hi = (lane >> 4) & 1;  // lane-half selects K sub-range / M-half
  // per-lane B source pointer: col 0 -> w_rel, col 1 -> w_time, others -> zeros
  const _Float16* wsel = (r == 0) ? Wr_s : ((r == 1) ? Wt_s : Z_s);

  v8f acc = {};
#pragma unroll
  for (int chunk = 0; chunk < 4; ++chunk) {
    const int kA = chunk * 32 + hi * 8;    // A elems 0..7 = K kA+i, 8..15 = K kA+16+i
    H16 a, b;
    a.p.lo = *(const v8h*)&A_s[r][kA];
    a.p.hi = *(const v8h*)&A_s[r][kA + 16];
    const int kB = chunk * 32 + hi * 8;    // B elems 0..7 = K kB+i, 8..15 = K kB+16+i
    b.p.lo = *(const v8h*)&wsel[kB];
    b.p.hi = *(const v8h*)&wsel[kB + 16];
    acc = __builtin_amdgcn_wmma_f32_16x16x32_f16(false, a.v, false, b.v, (short)0, acc,
                                                 false, false);
  }

  // D layout: lane = column, VGPR v = row (v + 8*hi). Column 0 = rel, column 1 = time.
  if (r < 2) {
    const float* d3 = (r == 0) ? rel_dot3 : time_dot3;
    const float  bs = (r == 0) ? b_rel[0] : b_time[0];
    float* op       = (r == 0) ? rel_logit : time_logit;
    const int rbase = hi * 8;
#pragma unroll
    for (int m = 0; m < 8; ++m) {
      int e = ebase + rbase + m;
      op[e] = acc[m] + d3[e] + bs;
    }
  }
}

__global__ void k_attn_init(int* mkr, int* mkt, float* sr, float* st, int n) {
  int i = blockIdx.x * blockDim.x + threadIdx.x;
  if (i < n) { mkr[i] = ORD_NEG_INF; mkt[i] = ORD_NEG_INF; sr[i] = 0.f; st[i] = 0.f; }
}

__global__ void k_attn_max(const float* rl, const float* tl, const int* src,
                           int* mkr, int* mkt) {
  int e = blockIdx.x * blockDim.x + threadIdx.x;
  if (e < EE) {
    int s = src[e];
    atomicMax(&mkr[s], f2ord(rl[e]));
    atomicMax(&mkt[s], f2ord(tl[e]));
  }
}

__global__ void k_attn_sum(const float* rl, const float* tl, const int* src,
                           const int* mkr, const int* mkt, float* sr, float* st) {
  int e = blockIdx.x * blockDim.x + threadIdx.x;
  if (e < EE) {
    int s = src[e];
    atomAddF(&sr[s], expf(rl[e] - ord2f(mkr[s])));
    atomAddF(&st[s], expf(tl[e] - ord2f(mkt[s])));
  }
}

__global__ void k_attn_feat(const float* rl, const float* tl, const int* src, const int* dst,
                            const int* mkr, const int* mkt, const float* sr, const float* st,
                            const float* ent_emb, float* out) {
  int i = blockIdx.x * blockDim.x + threadIdx.x;
  if (i < EE * DD) {
    int e = i / DD, d = i - e * DD;
    int s = src[e];
    float c = expf(rl[e] - ord2f(mkr[s])) / sr[s] +
              expf(tl[e] - ord2f(mkt[s])) / st[s];
    atomAddF(&out[(long)s * OC + DD + d], c * ent_emb[(long)dst[e] * DD + d]);
  }
}

__global__ void k_finalize(const float* ent_emb, float* out, int n) {
  int i = blockIdx.x * blockDim.x + threadIdx.x;
  if (i < n * DD) {
    int r = i / DD, d = i - r * DD;
    float* o = out + (long)r * OC;
    o[d] = fmaxf(ent_emb[i], 0.f);              // feats0 = relu(ent_emb)
    float f = fmaxf(o[DD + d], 0.f);            // feats = relu(aggregate)
    o[DD + d] = f;
    o[2 * DD + d] = f;                          // DEPTH=2 -> replicated
    o[3 * DD + d] = fmaxf(o[3 * DD + d], 0.f);  // time encoding relu
  }
}

extern "C" void kernel_launch(void* const* d_in, const int* in_sizes, int n_in,
                              void* d_out, int out_size, void* d_ws, size_t ws_size,
                              hipStream_t stream) {
  (void)in_sizes; (void)n_in; (void)out_size; (void)ws_size;
  const int*   adj_src   = (const int*)d_in[0];
  const int*   adj_dst   = (const int*)d_in[1];
  const int*   r_edge    = (const int*)d_in[2];
  const int*   r_rel     = (const int*)d_in[3];
  const int*   t_edge    = (const int*)d_in[4];
  const int*   t_time    = (const int*)d_in[5];
  const int*   ent_rows  = (const int*)d_in[6];
  const int*   ent_cols  = (const int*)d_in[7];
  const int*   rel_rows  = (const int*)d_in[8];
  const int*   rel_cols  = (const int*)d_in[9];
  const int*   time_rows = (const int*)d_in[10];
  const int*   time_cols = (const int*)d_in[11];
  const float* r_val     = (const float*)d_in[12];
  const float* ent_table = (const float*)d_in[13];
  const float* rel_table = (const float*)d_in[14];
  const float* time_table= (const float*)d_in[15];
  const float* w_rel     = (const float*)d_in[16];
  const float* b_rel     = (const float*)d_in[17];
  const float* w_time    = (const float*)d_in[18];
  const float* b_time    = (const float*)d_in[19];
  float* out = (float*)d_out;

  float* ws = (float*)d_ws;
  float* buf        = ws; ws += (size_t)EE * DD;   // 200 MB, reused rel -> time
  float* ent_emb    = ws; ws += (size_t)NN * DD;
  float* rel_emb    = ws; ws += (size_t)RR * DD;
  float* time_emb   = ws; ws += (size_t)TT * DD;
  float* rel_dot3   = ws; ws += EE;
  float* time_dot3  = ws; ws += EE;
  float* rel_logit  = ws; ws += EE;
  float* time_logit = ws; ws += EE;
  float* s_rel      = ws; ws += NN;
  float* s_time     = ws; ws += NN;
  int* cnt_ent  = (int*)ws; ws += NN;   // counts kept contiguous for one zero pass
  int* cnt_rel  = (int*)ws; ws += RR;
  int* cnt_time = (int*)ws; ws += TT;
  int* mk_rel   = (int*)ws; ws += NN;
  int* mk_time  = (int*)ws; ws += NN;

  const int B = 256;
  auto g = [](long n, int b) { return (int)((n + b - 1) / b); };

  // ---- init ----
  k_zero<<<g((long)NN * OC, B), B, 0, stream>>>(out, NN * OC);
  k_zero<<<g((long)NN * DD, B), B, 0, stream>>>(ent_emb, NN * DD);
  k_zero<<<g((long)RR * DD, B), B, 0, stream>>>(rel_emb, RR * DD);
  k_zero<<<g((long)TT * DD, B), B, 0, stream>>>(time_emb, TT * DD);
  k_zero<<<g((long)(NN + RR + TT), B), B, 0, stream>>>((float*)cnt_ent, NN + RR + TT);

  // ---- contextual embeddings (scatter-mean) ----
  k_count<<<g(NNZ_E, B), B, 0, stream>>>(ent_rows, NNZ_E, cnt_ent);
  k_count<<<g(NNZ_R, B), B, 0, stream>>>(rel_rows, NNZ_R, cnt_rel);
  k_count<<<g(NNZ_T, B), B, 0, stream>>>(time_rows, NNZ_T, cnt_time);
  k_ctx_scatter<<<g((long)NNZ_E * DD, B), B, 0, stream>>>(ent_rows, ent_cols, ent_table, ent_emb, NNZ_E);
  k_ctx_scatter<<<g((long)NNZ_R * DD, B), B, 0, stream>>>(rel_rows, rel_cols, rel_table, rel_emb, NNZ_R);
  k_ctx_scatter<<<g((long)NNZ_T * DD, B), B, 0, stream>>>(time_rows, time_cols, time_table, time_emb, NNZ_T);
  k_ctx_div<<<g((long)NN * DD, B), B, 0, stream>>>(ent_emb, cnt_ent, NN);
  k_ctx_div<<<g((long)RR * DD, B), B, 0, stream>>>(rel_emb, cnt_rel, RR);
  k_ctx_div<<<g((long)TT * DD, B), B, 0, stream>>>(time_emb, cnt_time, TT);

  // ---- rel phase (buf = rels_sum) ----
  k_zero<<<g((long)EE * DD, B), B, 0, stream>>>(buf, EE * DD);
  k_edge_scatter<<<g((long)EE * DD, B), B, 0, stream>>>(r_edge, r_rel, r_val, rel_emb, buf);
  k_norm_dot<<<g((long)EE * 32, B), B, 0, stream>>>(buf, w_rel, rel_dot3, 0, adj_src, out);

  // ---- time phase (buf = times_sum; also scatter time encoding into out[:,150:200]) ----
  k_zero<<<g((long)EE * DD, B), B, 0, stream>>>(buf, EE * DD);
  k_edge_scatter<<<g((long)EE * DD, B), B, 0, stream>>>(t_edge, t_time, r_val, time_emb, buf);
  k_norm_dot<<<g((long)EE * 32, B), B, 0, stream>>>(buf, w_time, time_dot3, 1, adj_src, out);

  // ---- per-edge logits via WMMA (both rel & time in one pass) ----
  k_logits_wmma<<<EE / 16, 32, 0, stream>>>(ent_emb, adj_src, adj_dst, w_rel, w_time,
                                            rel_dot3, time_dot3, b_rel, b_time,
                                            rel_logit, time_logit);

  // ---- segment softmax over adj_src + attention aggregate ----
  k_attn_init<<<g(NN, B), B, 0, stream>>>(mk_rel, mk_time, s_rel, s_time, NN);
  k_attn_max<<<g(EE, B), B, 0, stream>>>(rel_logit, time_logit, adj_src, mk_rel, mk_time);
  k_attn_sum<<<g(EE, B), B, 0, stream>>>(rel_logit, time_logit, adj_src, mk_rel, mk_time, s_rel, s_time);
  k_attn_feat<<<g((long)EE * DD, B), B, 0, stream>>>(rel_logit, time_logit, adj_src, adj_dst,
                                                     mk_rel, mk_time, s_rel, s_time, ent_emb, out);

  // ---- relu / replicate / finalize ----
  k_finalize<<<g((long)NN * DD, B), B, 0, stream>>>(ent_emb, out, NN);
}
